// AttentionalGCN_10952166605414
// MI455X (gfx1250) — compile-verified, matrix operand order checked by reading, not run on hardware
//
#include <hip/hip_runtime.h>
#include <hip/hip_bf16.h>

// ---------- types for CDNA5 WMMA ----------
typedef __attribute__((ext_vector_type(16))) __bf16 v16bf;
typedef __attribute__((ext_vector_type(8)))  __bf16 v8bf;
typedef __attribute__((ext_vector_type(8)))  float  v8f;

#define THREADS 256
#define NBLK 4   // 16x64 output strip per wave: 4 WMMA accumulators, A-frag reused 4x
static inline int cdiv(long long a, long long b) { return (int)((a + b - 1) / b); }

// ---------- helpers ----------
__device__ __forceinline__ unsigned fmap_u32(float f) {
    unsigned u = __float_as_uint(f);
    return (u & 0x80000000u) ? ~u : (u | 0x80000000u);  // order-preserving float->uint
}
__device__ __forceinline__ float funmap_u32(unsigned u) {
    return (u & 0x80000000u) ? __uint_as_float(u & 0x7FFFFFFFu) : __uint_as_float(~u);
}
__device__ __forceinline__ void edge_sd(const int* __restrict__ ei, int e, int E,
                                        int& s, int& d) {
    if (e < E) { s = ei[e]; d = ei[E + e]; }
    else       { s = d = e - E; }               // appended self-loops
}

// ---------- generic fill ----------
__global__ void fill_u32_kernel(unsigned* __restrict__ p, unsigned v, long long n) {
    long long i = (long long)blockIdx.x * blockDim.x + threadIdx.x;
    if (i < n) p[i] = v;
}

// ---------- fp32 -> bf16 convert ----------
__global__ void cvt_bf16_kernel(const float* __restrict__ src, __bf16* __restrict__ dst,
                                long long n) {
    long long i = (long long)blockIdx.x * blockDim.x + threadIdx.x;
    if (i < n) dst[i] = (__bf16)src[i];
}

// ---------- W [K,N] fp32 -> Wt [N,K] bf16 (so B-fragments load contiguously) ----------
__global__ void transpose_cvt_kernel(const float* __restrict__ W, __bf16* __restrict__ Wt,
                                     int K, int N) {
    long long i = (long long)blockIdx.x * blockDim.x + threadIdx.x;
    if (i >= (long long)K * N) return;
    int k = (int)(i % K);
    int n = (int)(i / K);
    Wt[i] = (__bf16)W[(long long)k * N + n];
}

// ---------- GEMM: C[M,N] = A[M,K] * B[K,N], A bf16 row-major, Bt = B^T bf16 [N,K] ----------
// One wave computes a 16x(16*NBLK) strip: A fragment loaded once per k-step, reused
// across NBLK v_wmma_f32_16x16x32_bf16 ops -> 4x WMMA density per A byte.
__global__ void gemm_bf16_wmma_kernel(const __bf16* __restrict__ A,
                                      const __bf16* __restrict__ Bt,
                                      float* __restrict__ C,
                                      int M, int Nn, int K) {
    const int wave = blockIdx.x * (blockDim.x >> 5) + (threadIdx.x >> 5);
    const int lane = threadIdx.x & 31;
    const int mtiles  = M >> 4;
    const int nstrips = Nn / (16 * NBLK);
    if (wave >= mtiles * nstrips) return;
    const int tm = wave / nstrips, tn = wave % nstrips;

    // ISA 16-bit A 16x32 layout: lane&15 = row/col, (lane>>4)*8 = K-half select
    const int rc   = lane & 15;
    const int ksel = (lane >> 4) * 8;
    const __bf16* arow = A + (long long)(tm * 16 + rc) * K + ksel;
    const __bf16* bcol[NBLK];
#pragma unroll
    for (int t = 0; t < NBLK; ++t)
        bcol[t] = Bt + (long long)((tn * NBLK + t) * 16 + rc) * K + ksel;

    v8f acc[NBLK] = {};
    for (int k0 = 0; k0 < K; k0 += 32) {
        const v8bf a_lo = *(const v8bf*)(arow + k0);
        const v8bf a_hi = *(const v8bf*)(arow + k0 + 16);
        v16bf a;
#pragma unroll
        for (int j = 0; j < 8; ++j) { a[j] = a_lo[j]; a[j + 8] = a_hi[j]; }
#pragma unroll
        for (int t = 0; t < NBLK; ++t) {
            const v8bf b_lo = *(const v8bf*)(bcol[t] + k0);
            const v8bf b_hi = *(const v8bf*)(bcol[t] + k0 + 16);
            v16bf b;
#pragma unroll
            for (int j = 0; j < 8; ++j) { b[j] = b_lo[j]; b[j + 8] = b_hi[j]; }
            acc[t] = __builtin_amdgcn_wmma_f32_16x16x32_bf16(
                /*neg_a=*/false, a, /*neg_b=*/false, b,
                /*c_mod=*/(short)0, acc[t], /*reuse_a=*/false, /*reuse_b=*/false);
        }
    }
    // C/D layout: col = lane&15, rows = (lane>>4)*8 + i
    const int mrow = tm * 16 + (lane >> 4) * 8;
#pragma unroll
    for (int t = 0; t < NBLK; ++t) {
        const int col = (tn * NBLK + t) * 16 + rc;
#pragma unroll
        for (int i = 0; i < 8; ++i)
            C[(long long)(mrow + i) * Nn + col] = acc[t][i];
    }
}

// ---------- per-(node,head) attention dots: one wave each ----------
__global__ void node_alpha_kernel(const float* __restrict__ h,
                                  const float* __restrict__ a_s,
                                  const float* __restrict__ a_d,
                                  float* __restrict__ os, float* __restrict__ od,
                                  int N, int H, int C) {
    const int wave = blockIdx.x * (blockDim.x >> 5) + (threadIdx.x >> 5);
    const int lane = threadIdx.x & 31;
    if (wave >= N * H) return;
    const int n = wave / H, hd = wave % H;
    const float* hp = h + (long long)n * H * C + hd * C;
    float ss = 0.f, sd = 0.f;
    for (int c = lane; c < C; c += 32) {
        float v = hp[c];
        ss += v * a_s[hd * C + c];
        sd += v * a_d[hd * C + c];
    }
#pragma unroll
    for (int off = 16; off; off >>= 1) {
        ss += __shfl_down(ss, off, 32);
        sd += __shfl_down(sd, off, 32);
    }
    if (lane == 0) { os[wave] = ss; od[wave] = sd; }
}

// ---------- edge pass 1: e = leaky_relu(as[src]+ad[dst]); segment max via mapped u32 ----------
__global__ void edge_max_kernel(const int* __restrict__ ei,
                                const float* __restrict__ as, const float* __restrict__ ad,
                                float* __restrict__ ebuf, unsigned* __restrict__ mmax,
                                int E, int N, int H) {
    const int idx = blockIdx.x * blockDim.x + threadIdx.x;
    if (idx >= (E + N) * H) return;
    const int e = idx / H, hd = idx - e * H;
    int s, d; edge_sd(ei, e, E, s, d);
    float v = as[s * H + hd] + ad[d * H + hd];
    v = v > 0.f ? v : 0.2f * v;                 // leaky_relu, NEG_SLOPE=0.2
    ebuf[idx] = v;
    atomicMax(&mmax[d * H + hd], fmap_u32(v));
}

// ---------- edge pass 2: ex = exp(e - max[dst]); segment sum ----------
__global__ void edge_expsum_kernel(const int* __restrict__ ei,
                                   float* __restrict__ ebuf,
                                   const unsigned* __restrict__ mmax,
                                   float* __restrict__ ssum,
                                   int E, int N, int H) {
    const int idx = blockIdx.x * blockDim.x + threadIdx.x;
    if (idx >= (E + N) * H) return;
    const int e = idx / H, hd = idx - e * H;
    int s, d; edge_sd(ei, e, E, s, d);
    (void)s;
    const float ex = __expf(ebuf[idx] - funmap_u32(mmax[d * H + hd]));
    ebuf[idx] = ex;
    atomicAdd(&ssum[d * H + hd], ex);
}

// ---------- edge pass 3: out[dst] += h[src] * (ex / sum[dst]); one wave per edge ----------
// h and out (41 MB each) are resident in MI455X's 192 MB L2 -> gather/scatter at L2 BW.
__global__ void edge_msg_kernel(const int* __restrict__ ei,
                                const float* __restrict__ h,
                                const float* __restrict__ ebuf,
                                const float* __restrict__ ssum,
                                float* __restrict__ out,
                                int E, int N, int H, int C) {
    const int wave = blockIdx.x * (blockDim.x >> 5) + (threadIdx.x >> 5);
    const int lane = threadIdx.x & 31;
    if (wave >= E + N) return;
    int s, d; edge_sd(ei, wave, E, s, d);
    const int F = H * C;
    const float* hp = h + (long long)s * F;
    float* op = out + (long long)d * F;
    for (int c = lane; c < F; c += 32) {        // coalesced 128B per instruction
        const int hd = c / C;
        const float alpha = ebuf[wave * H + hd] / ssum[d * H + hd];
        atomicAdd(&op[c], hp[c] * alpha);
    }
}

// ---------- epilogues ----------
__global__ void bias_relu_cvt_kernel(const float* __restrict__ acc,
                                     const float* __restrict__ bias,
                                     __bf16* __restrict__ xout, long long n, int F) {
    long long i = (long long)blockIdx.x * blockDim.x + threadIdx.x;
    if (i >= n) return;
    float v = acc[i] + bias[i % F];
    xout[i] = (__bf16)fmaxf(v, 0.f);
}
__global__ void bias_add_kernel(float* __restrict__ out, const float* __restrict__ bias,
                                long long n, int F) {
    long long i = (long long)blockIdx.x * blockDim.x + threadIdx.x;
    if (i >= n) return;
    out[i] += bias[i % F];
}

// ================= host side =================
extern "C" void kernel_launch(void* const* d_in, const int* in_sizes, int n_in,
                              void* d_out, int out_size, void* d_ws, size_t ws_size,
                              hipStream_t stream) {
    (void)n_in; (void)out_size; (void)ws_size;
    const int IN_DIM = 512, HID = 128, HEADS = 4, OUT_DIM = 256;
    const int F1 = HEADS * HID;                    // 512
    const int N  = in_sizes[0] / IN_DIM;           // 20000
    const int E  = in_sizes[1] / 2;                // 640000
    const int ET = E + N;                          // + self loops

    const float* x    = (const float*)d_in[0];
    const int*   ei   = (const int*)  d_in[1];     // harness: integer -> int32
    const float* W1   = (const float*)d_in[2];
    const float* as1w = (const float*)d_in[3];
    const float* ad1w = (const float*)d_in[4];
    const float* b1   = (const float*)d_in[5];
    const float* W2   = (const float*)d_in[6];
    const float* as2w = (const float*)d_in[7];
    const float* ad2w = (const float*)d_in[8];
    const float* b2   = (const float*)d_in[9];
    float* out = (float*)d_out;

    // workspace carve-up (256B aligned)
    char* wp = (char*)d_ws;
    auto carve = [&](long long bytes) -> void* {
        void* p = wp; wp += (bytes + 255) & ~255ll; return p;
    };
    __bf16*  xb   = (__bf16*) carve((long long)N * IN_DIM * 2);
    __bf16*  W1t  = (__bf16*) carve((long long)IN_DIM * F1 * 2);   // [F1, IN_DIM]
    __bf16*  W2t  = (__bf16*) carve((long long)F1 * OUT_DIM * 2);  // [OUT_DIM, F1]
    float*   h1   = (float*)  carve((long long)N * F1 * 4);
    float*   out1 = (float*)  carve((long long)N * F1 * 4);
    __bf16*  x2b  = (__bf16*) carve((long long)N * F1 * 2);
    float*   h2   = (float*)  carve((long long)N * OUT_DIM * 4);
    float*   as1  = (float*)  carve((long long)N * HEADS * 4);
    float*   ad1  = (float*)  carve((long long)N * HEADS * 4);
    unsigned* m1  = (unsigned*)carve((long long)N * HEADS * 4);
    float*   s1   = (float*)  carve((long long)N * HEADS * 4);
    float*   e1   = (float*)  carve((long long)ET * HEADS * 4);
    float*   as2  = (float*)  carve((long long)N * 4);
    float*   ad2  = (float*)  carve((long long)N * 4);
    unsigned* m2  = (unsigned*)carve((long long)N * 4);
    float*   s2   = (float*)  carve((long long)N * 4);
    float*   e2   = (float*)  carve((long long)ET * 4);

    // ---- precision conversion / weight transpose (bf16 WMMA operands) ----
    cvt_bf16_kernel<<<cdiv((long long)N * IN_DIM, THREADS), THREADS, 0, stream>>>(
        x, xb, (long long)N * IN_DIM);
    transpose_cvt_kernel<<<cdiv((long long)IN_DIM * F1, THREADS), THREADS, 0, stream>>>(
        W1, W1t, IN_DIM, F1);
    transpose_cvt_kernel<<<cdiv((long long)F1 * OUT_DIM, THREADS), THREADS, 0, stream>>>(
        W2, W2t, F1, OUT_DIM);

    // ================= layer 1 =================
    {
        const int strips = (N / 16) * (F1 / (16 * NBLK));
        gemm_bf16_wmma_kernel<<<cdiv(strips, 8), THREADS, 0, stream>>>(
            xb, W1t, h1, N, F1, IN_DIM);

        node_alpha_kernel<<<cdiv(N * HEADS, 8), THREADS, 0, stream>>>(
            h1, as1w, ad1w, as1, ad1, N, HEADS, HID);

        fill_u32_kernel<<<cdiv((long long)N * HEADS, THREADS), THREADS, 0, stream>>>(
            m1, 0u, (long long)N * HEADS);
        fill_u32_kernel<<<cdiv((long long)N * HEADS, THREADS), THREADS, 0, stream>>>(
            (unsigned*)s1, 0u, (long long)N * HEADS);
        fill_u32_kernel<<<cdiv((long long)N * F1, THREADS), THREADS, 0, stream>>>(
            (unsigned*)out1, 0u, (long long)N * F1);

        edge_max_kernel<<<cdiv((long long)ET * HEADS, THREADS), THREADS, 0, stream>>>(
            ei, as1, ad1, e1, m1, E, N, HEADS);
        edge_expsum_kernel<<<cdiv((long long)ET * HEADS, THREADS), THREADS, 0, stream>>>(
            ei, e1, m1, s1, E, N, HEADS);
        edge_msg_kernel<<<cdiv(ET, 8), THREADS, 0, stream>>>(
            ei, h1, e1, s1, out1, E, N, HEADS, HID);

        bias_relu_cvt_kernel<<<cdiv((long long)N * F1, THREADS), THREADS, 0, stream>>>(
            out1, b1, x2b, (long long)N * F1, F1);
    }

    // ================= layer 2 (heads=1, ch=256) =================
    {
        const int strips = (N / 16) * (OUT_DIM / (16 * NBLK));
        gemm_bf16_wmma_kernel<<<cdiv(strips, 8), THREADS, 0, stream>>>(
            x2b, W2t, h2, N, OUT_DIM, F1);

        node_alpha_kernel<<<cdiv(N, 8), THREADS, 0, stream>>>(
            h2, as2w, ad2w, as2, ad2, N, 1, OUT_DIM);

        fill_u32_kernel<<<cdiv((long long)N, THREADS), THREADS, 0, stream>>>(
            m2, 0u, (long long)N);
        fill_u32_kernel<<<cdiv((long long)N, THREADS), THREADS, 0, stream>>>(
            (unsigned*)s2, 0u, (long long)N);
        fill_u32_kernel<<<cdiv((long long)N * OUT_DIM, THREADS), THREADS, 0, stream>>>(
            (unsigned*)out, 0u, (long long)N * OUT_DIM);

        edge_max_kernel<<<cdiv((long long)ET, THREADS), THREADS, 0, stream>>>(
            ei, as2, ad2, e2, m2, E, N, 1);
        edge_expsum_kernel<<<cdiv((long long)ET, THREADS), THREADS, 0, stream>>>(
            ei, e2, m2, s2, E, N, 1);
        edge_msg_kernel<<<cdiv(ET, 8), THREADS, 0, stream>>>(
            ei, h2, e2, s2, out, E, N, 1, OUT_DIM);

        bias_add_kernel<<<cdiv((long long)N * OUT_DIM, THREADS), THREADS, 0, stream>>>(
            out, b2, (long long)N * OUT_DIM, OUT_DIM);
    }
}
